// MultiHeadAttention_41257455845612
// MI455X (gfx1250) — matrix operand order repaired and checked by reading, not compiled, not run on hardware
//
#include <hip/hip_runtime.h>

typedef __attribute__((ext_vector_type(16))) __bf16 v16bf;
typedef __attribute__((ext_vector_type(16))) unsigned short v16us;
typedef __attribute__((ext_vector_type(8))) float v8f;

#define SEQ    2048
#define DMODEL 1024
#define NHEADS 16
#define HDIM   64

__device__ __forceinline__ unsigned short f2bf(float f) {
  unsigned int u = __float_as_uint(f);
  u += 0x7FFFu + ((u >> 16) & 1u);  // round to nearest even
  return (unsigned short)(u >> 16);
}

struct U4x2 { uint4 a, b; };
__device__ __forceinline__ v16bf frag_from_chunks(uint4 lo, uint4 hi) {
  U4x2 t{lo, hi};
  return __builtin_bit_cast(v16bf, t);
}

// A-operand (ISA 7.12.2, 16-bit A 16x32): lane = half*16 + m.
//   half h holds K in [8h, 8h+8) and [16+8h, 16+8h+8)  -> two 16B chunks.
__device__ __forceinline__ v16bf load_frag_a(const unsigned short* p, int half) {
  uint4 lo = *(const uint4*)(p + half * 8);
  uint4 hi = *(const uint4*)(p + 16 + half * 8);
  return frag_from_chunks(lo, hi);
}

// B-operand (16-bit B 32x16): half h holds K in [8h, 8h+8) and
// [16+8h, 16+8h+8) -> two 16B chunks (p = K-major row of column n).
__device__ __forceinline__ v16bf load_frag_b(const unsigned short* p, int half) {
  uint4 lo = *(const uint4*)(p + half * 8);
  uint4 hi = *(const uint4*)(p + 16 + half * 8);
  return frag_from_chunks(lo, hi);
}

__device__ __forceinline__ v8f wmma_bf16(v16bf a, v16bf b, v8f c) {
  return __builtin_amdgcn_wmma_f32_16x16x32_bf16(false, a, false, b, (short)0, c,
                                                 false, false);
}

// ---------------------------------------------------------------- prep kernels
__global__ void __launch_bounds__(256) cvt_bf16_kernel(
    const float4* __restrict__ in, ushort4* __restrict__ out, int n4) {
  int i = blockIdx.x * 256 + threadIdx.x;
  if (i < n4) {
    float4 v = in[i];
    ushort4 o;
    o.x = f2bf(v.x); o.y = f2bf(v.y); o.z = f2bf(v.z); o.w = f2bf(v.w);
    out[i] = o;
  }
}

// W (K x N fp32, row-major) -> Wt (N x K bf16, row-major)
__global__ void __launch_bounds__(256) transpose_bf16_kernel(
    const float* __restrict__ in, unsigned short* __restrict__ out,
    int Kdim, int Ndim) {
  __shared__ unsigned short tile[32][33];
  int n0 = blockIdx.x * 32, k0 = blockIdx.y * 32;
  int tx = threadIdx.x & 31, ty = threadIdx.x >> 5;  // 32 x 8
#pragma unroll
  for (int r = 0; r < 32; r += 8)
    tile[ty + r][tx] = f2bf(in[(size_t)(k0 + ty + r) * Ndim + n0 + tx]);
  __syncthreads();
#pragma unroll
  for (int r = 0; r < 32; r += 8)
    out[(size_t)(n0 + ty + r) * Kdim + k0 + tx] = tile[tx][ty + r];
}

// ------------------------------------------------------------------- GEMM (TN)
// Out = (X(MxK) @ Wt(NxK)^T + bias) * oscale.  128 thr = 4 waves; wave 32x64.
// OUT_MODE: 0 = fp32 row-major, 1 = bf16 row-major, 2 = bf16 V-transposed
//           (Vt[(n*2 + batch)*SEQ + seq], n = h*64+d)
template <int OUT_MODE>
__global__ void __launch_bounds__(128) gemm_tn_kernel(
    const unsigned short* __restrict__ X, const unsigned short* __restrict__ Wt,
    const float* __restrict__ bias, void* __restrict__ OutP,
    int M, int N, int K, float oscale) {
  int lane = threadIdx.x & 31;
  int half = lane >> 4, mn = lane & 15;
  int wave = threadIdx.x >> 5;
  int m0 = (blockIdx.x * 4 + wave) * 32;
  int n0 = blockIdx.y * 64;

  const unsigned short* xr0 = X + (size_t)(m0 + mn) * K;
  const unsigned short* xr1 = X + (size_t)(m0 + 16 + mn) * K;
  const unsigned short* wr = Wt + (size_t)(n0 + mn) * K;

  v8f acc[2][4] = {};
  for (int k0 = 0; k0 < K; k0 += 32) {
    v16bf a0 = load_frag_a(xr0 + k0, half);
    v16bf a1 = load_frag_a(xr1 + k0, half);
#pragma unroll
    for (int t = 0; t < 4; ++t) {
      v16bf b = load_frag_b(wr + (size_t)t * 16 * K + k0, half);
      acc[0][t] = wmma_bf16(a0, b, acc[0][t]);
      acc[1][t] = wmma_bf16(a1, b, acc[1][t]);
    }
  }

#pragma unroll
  for (int t = 0; t < 4; ++t) {
    int n = n0 + t * 16 + mn;
    float bval = bias[n];
#pragma unroll
    for (int i = 0; i < 2; ++i)
#pragma unroll
      for (int r = 0; r < 8; ++r) {
        int m = m0 + i * 16 + r + half * 8;
        float val = (acc[i][t][r] + bval) * oscale;
        if constexpr (OUT_MODE == 0) {
          ((float*)OutP)[(size_t)m * N + n] = val;
        } else if constexpr (OUT_MODE == 1) {
          ((unsigned short*)OutP)[(size_t)m * N + n] = f2bf(val);
        } else {
          int bb = m >> 11, s = m & (SEQ - 1);
          ((unsigned short*)OutP)[((size_t)n * 2 + bb) * SEQ + s] = f2bf(val);
        }
      }
  }
}

// ---------------------------------------------------------------- attention
// One 32-key chunk of flash attention.  Scale is pre-folded into Q.
// Row sums come from an extra WMMA against an all-ones B (no shuffle reduce).
template <bool MASKED>
__device__ __forceinline__ void attn_chunk(
    int k0, int q0, int half, int mn,
    const unsigned short* __restrict__ Kb,
    const unsigned short* __restrict__ vr0, const unsigned short* __restrict__ vr1,
    const unsigned short* __restrict__ vr2, const unsigned short* __restrict__ vr3,
    v16bf qa0, v16bf qa1, v16bf ones,
    float (*__restrict__ Pslice)[36],
    float* __restrict__ mrow, float* __restrict__ lrow, v8f* __restrict__ acc) {
  // S = Q @ K^T: two 16x16 f32 tiles
  v8f s[2];
#pragma unroll
  for (int j = 0; j < 2; ++j) {
    const unsigned short* krow = Kb + (size_t)(k0 + j * 16 + mn) * DMODEL;
    v16bf bk0 = load_frag_b(krow, half);
    v16bf bk1 = load_frag_b(krow + 32, half);
    v8f z = {};
    s[j] = wmma_bf16(qa0, bk0, z);
    s[j] = wmma_bf16(qa1, bk1, s[j]);
  }

  // online-softmax max phase (row = one 16-lane group); write exp(P) to LDS
  float alpha[8];
#pragma unroll
  for (int r = 0; r < 8; ++r) {
    if constexpr (MASKED) {
      int qi = q0 + r + half * 8;
#pragma unroll
      for (int j = 0; j < 2; ++j) {
        int ke = k0 + j * 16 + mn;
        s[j][r] = (ke <= qi) ? s[j][r] : -1e30f;
      }
    }
    float mx = fmaxf(s[0][r], s[1][r]);
#pragma unroll
    for (int d = 1; d < 16; d <<= 1) mx = fmaxf(mx, __shfl_xor(mx, d, 32));
    float mnew = fmaxf(mrow[r], mx);
    alpha[r] = __expf(mrow[r] - mnew);
    mrow[r] = mnew;
    Pslice[r + half * 8][mn] = __expf(s[0][r] - mnew);
    Pslice[r + half * 8][16 + mn] = __expf(s[1][r] - mnew);
  }
#pragma unroll
  for (int t = 0; t < 4; ++t)
#pragma unroll
    for (int r = 0; r < 8; ++r) acc[t][r] *= alpha[r];

  __asm__ volatile("s_wait_dscnt 0" ::: "memory");  // same-wave cross-lane LDS RAW

  // reshape P (C layout fp32) -> A layout bf16
  const float* prow = &Pslice[mn][0];
  v16us pu;
#pragma unroll
  for (int e = 0; e < 8; ++e) pu[e] = f2bf(prow[half * 8 + e]);
#pragma unroll
  for (int e = 0; e < 8; ++e) pu[8 + e] = f2bf(prow[16 + half * 8 + e]);
  v16bf pa = __builtin_bit_cast(v16bf, pu);

  // row sums = P @ ones: each lane's 8 accum VGPRs hold its 8 row sums
  v8f z = {};
  v8f sums = wmma_bf16(pa, ones, z);

  // O += P(16x32) @ V(32x64): V fragments contiguous via head-transposed Vt
  acc[0] = wmma_bf16(pa, load_frag_b(vr0 + k0, half), acc[0]);
  acc[1] = wmma_bf16(pa, load_frag_b(vr1 + k0, half), acc[1]);
  acc[2] = wmma_bf16(pa, load_frag_b(vr2 + k0, half), acc[2]);
  acc[3] = wmma_bf16(pa, load_frag_b(vr3 + k0, half), acc[3]);

#pragma unroll
  for (int r = 0; r < 8; ++r) lrow[r] = lrow[r] * alpha[r] + sums[r];
}

// Flash attention: one wave per (batch, head, 16-row q tile); key chunks of 32.
// Q,K: [b*SEQ + s][h*64 + d] bf16 (Q pre-scaled by 1/8).  Vt: [(h*64+d)*2 + b][s].
__global__ void __launch_bounds__(128) attn_kernel(
    const unsigned short* __restrict__ Q, const unsigned short* __restrict__ Kd,
    const unsigned short* __restrict__ Vt, unsigned short* __restrict__ Ctx) {
  __shared__ float Pld[4][16][36];
  int lane = threadIdx.x & 31;
  int half = lane >> 4, mn = lane & 15;
  int wave = threadIdx.x >> 5;
  int b = blockIdx.x >> 4;
  int h = blockIdx.x & 15;
  int q0 = (blockIdx.y * 4 + wave) * 16;

  size_t qkbase = (size_t)b * SEQ * DMODEL + (size_t)h * HDIM;
  const unsigned short* Qb = Q + qkbase;
  const unsigned short* Kb = Kd + qkbase;

  const unsigned short* qrow = Qb + (size_t)(q0 + mn) * DMODEL;
  v16bf qa0 = load_frag_a(qrow, half);
  v16bf qa1 = load_frag_a(qrow + 32, half);

  const unsigned short* vr[4];
#pragma unroll
  for (int t = 0; t < 4; ++t)
    vr[t] = Vt + ((size_t)(h * HDIM + t * 16 + mn) * 2 + b) * SEQ;

  v16us ou;
#pragma unroll
  for (int e = 0; e < 16; ++e) ou[e] = 0x3F80;  // bf16 1.0
  v16bf ones = __builtin_bit_cast(v16bf, ou);

  float mrow[8], lrow[8];
#pragma unroll
  for (int r = 0; r < 8; ++r) { mrow[r] = -1e30f; lrow[r] = 0.f; }
  v8f acc[4] = {};

  // chunks with k0+31 <= q0 need no causal mask; exactly one masked chunk remains
  int kfull_end = (q0 >= 31) ? ((((q0 - 31) >> 5) << 5) + 32) : 0;
  int k0 = 0;
  for (; k0 < kfull_end; k0 += 32)
    attn_chunk<false>(k0, q0, half, mn, Kb, vr[0], vr[1], vr[2], vr[3],
                      qa0, qa1, ones, Pld[wave], mrow, lrow, acc);
  for (; k0 < q0 + 16; k0 += 32)
    attn_chunk<true>(k0, q0, half, mn, Kb, vr[0], vr[1], vr[2], vr[3],
                     qa0, qa1, ones, Pld[wave], mrow, lrow, acc);

#pragma unroll
  for (int r = 0; r < 8; ++r) lrow[r] = 1.0f / lrow[r];
#pragma unroll
  for (int t = 0; t < 4; ++t)
#pragma unroll
    for (int r = 0; r < 8; ++r) {
      size_t qi = (size_t)q0 + r + half * 8;
      Ctx[((size_t)b * SEQ + qi) * DMODEL + h * HDIM + t * 16 + mn] =
          f2bf(acc[t][r] * lrow[r]);
    }
}

// ------------------------------------------------------------------- launcher
extern "C" void kernel_launch(void* const* d_in, const int* in_sizes, int n_in,
                              void* d_out, int out_size, void* d_ws, size_t ws_size,
                              hipStream_t stream) {
  const float* x  = (const float*)d_in[0];
  const float* Wq = (const float*)d_in[1];
  const float* bq = (const float*)d_in[2];
  const float* Wk = (const float*)d_in[3];
  const float* bk = (const float*)d_in[4];
  const float* Wv = (const float*)d_in[5];
  const float* bv = (const float*)d_in[6];
  const float* Wo = (const float*)d_in[7];
  const float* bo = (const float*)d_in[8];

  const int M = 2 * SEQ;  // 4096 rows
  const size_t DD = (size_t)DMODEL * DMODEL;

  unsigned short* Xbf = (unsigned short*)d_ws;
  unsigned short* Wqt = Xbf + (size_t)M * DMODEL;
  unsigned short* Wkt = Wqt + DD;
  unsigned short* Wvt = Wkt + DD;
  unsigned short* Wot = Wvt + DD;
  unsigned short* Qw  = Wot + DD;
  unsigned short* Kw  = Qw + (size_t)M * DMODEL;
  unsigned short* Vtw = Kw + (size_t)M * DMODEL;
  unsigned short* Cw  = Vtw + (size_t)M * DMODEL;  // total 48 MB

  int n4 = (M * DMODEL) / 4;
  cvt_bf16_kernel<<<n4 / 256, 256, 0, stream>>>((const float4*)x, (ushort4*)Xbf, n4);

  dim3 tgrid(DMODEL / 32, DMODEL / 32);
  transpose_bf16_kernel<<<tgrid, 256, 0, stream>>>(Wq, Wqt, DMODEL, DMODEL);
  transpose_bf16_kernel<<<tgrid, 256, 0, stream>>>(Wk, Wkt, DMODEL, DMODEL);
  transpose_bf16_kernel<<<tgrid, 256, 0, stream>>>(Wv, Wvt, DMODEL, DMODEL);
  transpose_bf16_kernel<<<tgrid, 256, 0, stream>>>(Wo, Wot, DMODEL, DMODEL);

  dim3 ggrid(M / 128, DMODEL / 64);
  // fold 1/sqrt(64) into Q projection: softmax(QK^T/8) == softmax((Q/8)K^T)
  gemm_tn_kernel<1><<<ggrid, 128, 0, stream>>>(Xbf, Wqt, bq, Qw, M, DMODEL, DMODEL, 0.125f);
  gemm_tn_kernel<1><<<ggrid, 128, 0, stream>>>(Xbf, Wkt, bk, Kw, M, DMODEL, DMODEL, 1.0f);
  gemm_tn_kernel<2><<<ggrid, 128, 0, stream>>>(Xbf, Wvt, bv, Vtw, M, DMODEL, DMODEL, 1.0f);

  attn_kernel<<<dim3(2 * NHEADS, SEQ / 64), 128, 0, stream>>>(Qw, Kw, Vtw, Cw);

  gemm_tn_kernel<0><<<ggrid, 128, 0, stream>>>(Cw, Wot, bo, d_out, M, DMODEL, DMODEL, 1.0f);
}
